// Codebook_68882685493533
// MI455X (gfx1250) — matrix operand order repaired and checked by reading, not compiled, not run on hardware
//
#include <hip/hip_runtime.h>
#include <hip/hip_bf16.h>
#include <stdint.h>

#define N_ROWS   16384
#define DIM      256
#define K_CODES  8192
#define M_TILE   64
#define K_CHUNK  32
#define KSPLIT   2
#define CHUNK_BYTES (K_CHUNK * DIM * 2)      // 16 KB per (hi|lo) array per chunk
#define BUF_STRIDE  32768                    // hi(16K) + lo(16K) per buffer

typedef __bf16 bfrag __attribute__((ext_vector_type(16)));
typedef float  v8f   __attribute__((ext_vector_type(8)));
typedef unsigned int v8u __attribute__((ext_vector_type(8)));

static __device__ __forceinline__ unsigned short f2bf(float x) {
    unsigned u = __float_as_uint(x);
    unsigned r = u + 0x7FFFu + ((u >> 16) & 1u);   // round-to-nearest-even
    return (unsigned short)(r >> 16);
}
static __device__ __forceinline__ float bf2f(unsigned short h) {
    return __uint_as_float(((unsigned)h) << 16);
}
// Assemble a 16 x bf16 WMMA fragment from two 8-element (16B) LDS segments.
static __device__ __forceinline__ bfrag ld_frag(const unsigned short* p0,
                                                const unsigned short* p1) {
    uint4 a = *(const uint4*)p0;
    uint4 b = *(const uint4*)p1;
    v8u t;
    t[0] = a.x; t[1] = a.y; t[2] = a.z; t[3] = a.w;
    t[4] = b.x; t[5] = b.y; t[6] = b.z; t[7] = b.w;
    return __builtin_bit_cast(bfrag, t);
}

// CDNA5 async global->LDS copy (16B per lane), tracked by ASYNCcnt.
static __device__ __forceinline__ void async_b128(unsigned lds, const void* base,
                                                  unsigned voff) {
    asm volatile("global_load_async_to_lds_b128 %0, %1, %2"
                 :: "v"(lds), "v"(voff), "s"(base) : "memory");
}
static __device__ __forceinline__ void wait_async_le8(void) {
    asm volatile("s_wait_asynccnt 0x8" ::: "memory");
}
static __device__ __forceinline__ void wait_async_0(void) {
    asm volatile("s_wait_asynccnt 0x0" ::: "memory");
}

// Stage one 32-code chunk (hi+lo bf16, 32 KB) into LDS: 8 async b128 per thread.
static __device__ __forceinline__ void stage_chunk_async(
        unsigned ldsHi, const unsigned short* gHi, const unsigned short* gLo,
        unsigned gbyte, int tid) {
    unsigned t16 = (unsigned)tid * 16u;
#pragma unroll
    for (int r = 0; r < 4; ++r) {
        unsigned o = t16 + (unsigned)r * 4096u;
        async_b128(ldsHi + o,          gHi, gbyte + o);
        async_b128(ldsHi + 16384u + o, gLo, gbyte + o);
    }
}

// ---------------------------------------------------------------------------
// Kernel 1: (b,c,h,w) -> (N, c) transpose producing encoded_flat.
// ---------------------------------------------------------------------------
__global__ void vq_transpose_in(const float* __restrict__ z, float* __restrict__ enc) {
    __shared__ float tile[M_TILE][DIM + 1];
    const int tid = threadIdx.x;
    const int n0  = blockIdx.x * M_TILE;
    const int b   = n0 >> 10;
    const int hw0 = n0 & 1023;
    const float* zb = z + (size_t)b * DIM * 1024 + hw0;
#pragma unroll 4
    for (int j = 0; j < 64; ++j) {
        int nn = tid & 63;
        int cc = (tid >> 6) + j * 4;
        tile[nn][cc] = zb[(size_t)cc * 1024 + nn];
    }
    __syncthreads();
    float* er = enc + (size_t)n0 * DIM;
#pragma unroll 4
    for (int j = 0; j < 64; ++j) {
        er[(size_t)j * DIM + tid] = tile[j][tid];
    }
}

// ---------------------------------------------------------------------------
// Kernel 2: e_sq[k] = ||embedding[k]||^2 (exact fp32)
// ---------------------------------------------------------------------------
__global__ void vq_esq(const float* __restrict__ emb, float* __restrict__ esq) {
    int k = blockIdx.x * blockDim.x + threadIdx.x;
    if (k >= K_CODES) return;
    const float4* r = (const float4*)(emb + (size_t)k * DIM);
    float s = 0.f;
#pragma unroll 8
    for (int i = 0; i < DIM / 4; ++i) {
        float4 v = r[i];
        s += v.x * v.x + v.y * v.y + v.z * v.z + v.w * v.w;
    }
    esq[k] = s;
}

// ---------------------------------------------------------------------------
// Kernel 3: split embedding fp32 -> bf16 (hi, lo) arrays in workspace, once.
// ---------------------------------------------------------------------------
__global__ void vq_split(const float* __restrict__ emb,
                         unsigned short* __restrict__ eHi,
                         unsigned short* __restrict__ eLo) {
    int t = blockIdx.x * blockDim.x + threadIdx.x;     // one per 4 floats
    float4 v = ((const float4*)emb)[t];
    unsigned short h0 = f2bf(v.x), h1 = f2bf(v.y), h2 = f2bf(v.z), h3 = f2bf(v.w);
    ushort4 hi; hi.x = h0; hi.y = h1; hi.z = h2; hi.w = h3;
    ushort4 lo;
    lo.x = f2bf(v.x - bf2f(h0)); lo.y = f2bf(v.y - bf2f(h1));
    lo.z = f2bf(v.z - bf2f(h2)); lo.w = f2bf(v.w - bf2f(h3));
    ((ushort4*)eHi)[t] = hi;
    ((ushort4*)eLo)[t] = lo;
}

// ---------------------------------------------------------------------------
// Kernel 4: WMMA distance GEMM + running argmin.
//  - A fragments (invariant over K) hoisted to registers after one LDS pass.
//  - B chunks streamed bf16 hi/lo via async global->LDS, double buffered.
// ---------------------------------------------------------------------------
__global__ __launch_bounds__(256, 1) void vq_argmin(
        const float* __restrict__ enc,
        const unsigned short* __restrict__ eHi, const unsigned short* __restrict__ eLo,
        const float* __restrict__ esq,
        float* __restrict__ candV, int* __restrict__ candI) {
    __shared__ __align__(16) unsigned char pool[2 * BUF_STRIDE];  // 64 KB, reused
    __shared__ float sRedV[8][16];
    __shared__ int   sRedI[8][16];

    const int tid   = threadIdx.x;
    const int n0    = blockIdx.x * M_TILE;
    const int kbase = blockIdx.y * (K_CODES / KSPLIT);
    const int kend  = kbase + K_CODES / KSPLIT;

    const int lane = tid & 31;
    const int wave = tid >> 5;
    const int rg   = wave & 3;   // row group (16 rows)
    const int ch   = wave >> 2;  // code half of chunk (16 codes)
    const int half = lane >> 4;
    const int lcol = lane & 15;

    // ---- Phase 1: stage A rows (fp32 -> hi/lo bf16) into the pool ----
    unsigned short* sAhi = (unsigned short*)pool;            // 32 KB
    unsigned short* sAlo = (unsigned short*)(pool + 32768);  // 32 KB
    {
        const float* ar = enc + (size_t)n0 * DIM;
#pragma unroll 4
        for (int i = tid; i < M_TILE * DIM; i += 256) {
            float x = ar[i];
            unsigned short h = f2bf(x);
            sAhi[i] = h;
            sAlo[i] = f2bf(x - bf2f(h));
        }
    }
    __syncthreads();

    // ---- Hoist all A fragments into registers (invariant over K loop) ----
    bfrag aHi[8], aLo[8];
    {
        const int arow = (rg * 16 + lcol) * DIM;
#pragma unroll
        for (int kc = 0; kc < 8; ++kc) {
            int s0 = kc * 32 + (half ? 16 : 0);
            aHi[kc] = ld_frag(&sAhi[arow + s0], &sAhi[arow + s0 + 8]);
            aLo[kc] = ld_frag(&sAlo[arow + s0], &sAlo[arow + s0 + 8]);
        }
    }
    __syncthreads();   // pool free for reuse as B double-buffer

    const unsigned poolBase = (unsigned)(size_t)(void*)pool;  // LDS byte offset

    float bestV[8];
    int   bestI[8];
#pragma unroll
    for (int v = 0; v < 8; ++v) { bestV[v] = 3.4e38f; bestI[v] = 0; }

    const int bcode = (ch * 16 + lcol) * DIM;   // element offset of lane's code row

    // Prologue: async-stage first chunk into buffer 0.
    stage_chunk_async(poolBase, eHi, eLo, (unsigned)kbase * DIM * 2u, tid);

    int parity = 0;
    for (int k0 = kbase; k0 < kend; k0 += K_CHUNK) {
        const bool has_next = (k0 + K_CHUNK) < kend;
        if (has_next) {
            stage_chunk_async(poolBase + (unsigned)(parity ^ 1) * BUF_STRIDE,
                              eHi, eLo, (unsigned)(k0 + K_CHUNK) * DIM * 2u, tid);
            wait_async_le8();   // in-order: current chunk's 8 copies complete
        } else {
            wait_async_0();
        }
        __syncthreads();        // all waves' copies of chunk k0 visible

        const unsigned short* bHi =
            (const unsigned short*)(pool + (size_t)parity * BUF_STRIDE);
        const unsigned short* bLo = bHi + (CHUNK_BYTES / 2);

        v8f acc1 = {};   // hi*hi chain
        v8f acc2 = {};   // cross-term chain (independent for ILP)
#pragma unroll
        for (int kc = 0; kc < DIM / 32; ++kc) {
            int t0 = kc * 32 + half * 16;
            bfrag bhi = ld_frag(&bHi[bcode + t0], &bHi[bcode + t0 + 8]);
            bfrag blo = ld_frag(&bLo[bcode + t0], &bLo[bcode + t0 + 8]);
            acc1 = __builtin_amdgcn_wmma_f32_16x16x32_bf16(
                       false, aHi[kc], false, bhi, (short)0, acc1, false, false);
            acc2 = __builtin_amdgcn_wmma_f32_16x16x32_bf16(
                       false, aHi[kc], false, blo, (short)0, acc2, false, false);
            acc2 = __builtin_amdgcn_wmma_f32_16x16x32_bf16(
                       false, aLo[kc], false, bhi, (short)0, acc2, false, false);
        }

        const int code = k0 + ch * 16 + lcol;
        const float eq = esq[code];
#pragma unroll
        for (int v = 0; v < 8; ++v) {
            float d = fmaf(-2.0f, acc1[v] + acc2[v], eq);   // ||e||^2 - 2 x.e
            if (d < bestV[v]) { bestV[v] = d; bestI[v] = code; }
        }
        __syncthreads();        // chunk consumed; next iter may overwrite buffer
        parity ^= 1;
    }

    // Reduce (min, argmin) across the 16 lanes of each half-wave.
#pragma unroll
    for (int v = 0; v < 8; ++v) {
        float bv = bestV[v]; int bi = bestI[v];
#pragma unroll
        for (int m = 1; m < 16; m <<= 1) {
            float ov = __shfl_xor(bv, m, 32);
            int   oi = __shfl_xor(bi, m, 32);
            if (ov < bv || (ov == bv && oi < bi)) { bv = ov; bi = oi; }
        }
        if (lcol == 0) { sRedV[wave][v + 8 * half] = bv; sRedI[wave][v + 8 * half] = bi; }
    }
    __syncthreads();

    if (tid < M_TILE) {
        int rgo = tid >> 4, lr = tid & 15;
        float v0 = sRedV[rgo][lr];     int i0 = sRedI[rgo][lr];
        float v1 = sRedV[rgo + 4][lr]; int i1 = sRedI[rgo + 4][lr];
        bool take1 = (v1 < v0) || (v1 == v0 && i1 < i0);
        size_t o = (size_t)blockIdx.y * N_ROWS + n0 + tid;
        candV[o] = take1 ? v1 : v0;
        candI[o] = take1 ? i1 : i0;
    }
}

// ---------------------------------------------------------------------------
// Kernel 5: merge K-split candidates, gather rows, write indices/qf/qt.
// ---------------------------------------------------------------------------
__global__ void vq_gather(const float* __restrict__ emb,
                          const float* __restrict__ candV, const int* __restrict__ candI,
                          float* __restrict__ outIdx, float* __restrict__ qf,
                          float* __restrict__ qt) {
    __shared__ float tile[M_TILE][DIM + 1];
    __shared__ int   sIdx[M_TILE];
    const int tid = threadIdx.x;
    const int n0  = blockIdx.x * M_TILE;

    if (tid < M_TILE) {
        int n = n0 + tid;
        float v0 = candV[n];          int i0 = candI[n];
        float v1 = candV[N_ROWS + n]; int i1 = candI[N_ROWS + n];
        bool take1 = (v1 < v0) || (v1 == v0 && i1 < i0);
        int bi = take1 ? i1 : i0;
        sIdx[tid] = bi;
        outIdx[n] = (float)bi;
    }
    __syncthreads();

#pragma unroll 4
    for (int j = 0; j < M_TILE; ++j) {
        int idx = sIdx[j];
        float v = emb[(size_t)idx * DIM + tid];
        qf[(size_t)(n0 + j) * DIM + tid] = v;     // quantized_flat == lookup (STE)
        tile[j][tid] = v;
    }
    __syncthreads();

    const int b = n0 >> 10, hw0 = n0 & 1023;
    float* qb = qt + (size_t)b * DIM * 1024 + hw0;
#pragma unroll 4
    for (int j = 0; j < 64; ++j) {
        int nn = tid & 63;
        int cc = (tid >> 6) + j * 4;
        qb[(size_t)cc * 1024 + nn] = tile[nn][cc];
    }
}

// ---------------------------------------------------------------------------
extern "C" void kernel_launch(void* const* d_in, const int* in_sizes, int n_in,
                              void* d_out, int out_size, void* d_ws, size_t ws_size,
                              hipStream_t stream) {
    const float* z   = (const float*)d_in[0];   // (16,256,32,32)
    const float* emb = (const float*)d_in[1];   // (8192,256)

    float* out  = (float*)d_out;
    float* enc  = out;                               // encoded_flat   16384*256
    float* qf   = enc + (size_t)N_ROWS * DIM;        // quantized_flat 16384*256
    float* oidx = qf + (size_t)N_ROWS * DIM;         // codebook_indices 16384
    float* qt   = oidx + N_ROWS;                     // quantized      16384*256

    float* esq   = (float*)d_ws;                           //  8192 f32
    float* candV = esq + K_CODES;                          //  2*16384 f32
    int*   candI = (int*)(candV + KSPLIT * N_ROWS);        //  2*16384 i32
    unsigned short* eHi = (unsigned short*)(candI + KSPLIT * N_ROWS);  // 4 MB
    unsigned short* eLo = eHi + (size_t)K_CODES * DIM;                 // 4 MB

    vq_transpose_in<<<N_ROWS / M_TILE, 256, 0, stream>>>(z, enc);
    vq_esq<<<K_CODES / 256, 256, 0, stream>>>(emb, esq);
    vq_split<<<(K_CODES * DIM / 4) / 256, 256, 0, stream>>>(emb, eHi, eLo);
    dim3 grid(N_ROWS / M_TILE, KSPLIT);
    vq_argmin<<<grid, 256, 0, stream>>>(enc, eHi, eLo, esq, candV, candI);
    vq_gather<<<N_ROWS / M_TILE, 256, 0, stream>>>(emb, candV, candI, oidx, qf, qt);
}
